// KernelUpdateHead_28621662061055
// MI455X (gfx1250) — matrix operand order repaired and checked by reading, not compiled, not run on hardware
//
#include <hip/hip_runtime.h>
#include <hip/hip_bf16.h>

// ---------------------------------------------------------------------------
// Problem constants
// ---------------------------------------------------------------------------
#define BB   2
#define NN   100
#define CC   256
#define HWSZ 40000            // H*W
#define NHH  8
#define DH   32               // CC / NHH
#define FFD  2048
#define NCL  133
#define BN   200              // BB*NN

constexpr int NT     = 7;     // ceil(100/16) n-tiles
constexpr int CT2    = 8;     // 256/32 c-tile *pairs*
constexpr int KSPLIT = 25;    // HW split for pooling
constexpr int KCH    = HWSZ / KSPLIT;   // 1600, multiple of 32
constexpr int HWT2   = HWSZ / 32;       // 1250 hw-tile pairs

// ---------------------------------------------------------------------------
// WMMA types / helpers (CDNA5 gfx1250, wave32)
// ---------------------------------------------------------------------------
typedef __attribute__((ext_vector_type(16))) __bf16 v16bf;
typedef __attribute__((ext_vector_type(8)))  float  v8f;

__device__ __forceinline__ __bf16 bf_zero() {
  return __builtin_bit_cast(__bf16, (unsigned short)0);
}
__device__ __forceinline__ __bf16 bf_hard(float m) {   // (sigmoid(m) > 0.5) == (m > 0)
  return __builtin_bit_cast(__bf16, (unsigned short)(m > 0.f ? 0x3F80u : 0u));
}

#if __has_builtin(__builtin_amdgcn_cvt_pk_bf16_f32)
// Hardware packed conversion: 2 floats -> 2 bf16 in one VALU op.
__device__ __forceinline__ void cvt2(v16bf& d, int i, float a, float b) {
  auto r = __builtin_amdgcn_cvt_pk_bf16_f32(a, b);
  unsigned u = __builtin_bit_cast(unsigned, r);
  d[i]     = __builtin_bit_cast(__bf16, (unsigned short)(u & 0xFFFFu));
  d[i + 1] = __builtin_bit_cast(__bf16, (unsigned short)(u >> 16));
}
#else
// Native fptrunc: clang lowers f32->bf16 inline; on gfx1250 (native bf16 VALU)
// this should select the hardware convert rather than the add/shift sequence.
__device__ __forceinline__ void cvt2(v16bf& d, int i, float a, float b) {
  d[i]     = static_cast<__bf16>(a);
  d[i + 1] = static_cast<__bf16>(b);
}
#endif

__device__ __forceinline__ void cvt4(v16bf& d, int i, const float4 f) {
  cvt2(d, i, f.x, f.y); cvt2(d, i + 2, f.z, f.w);
}

// A-fragment (16x32 bf16): lane half h holds K = h*16 + e (16 consecutive).
// p points at (row, k0 + half*16).
__device__ __forceinline__ v16bf load_a_frag(const float* p, bool valid) {
  v16bf a;
  if (valid) {
    cvt4(a, 0,  *reinterpret_cast<const float4*>(p));
    cvt4(a, 4,  *reinterpret_cast<const float4*>(p + 4));
    cvt4(a, 8,  *reinterpret_cast<const float4*>(p + 8));
    cvt4(a, 12, *reinterpret_cast<const float4*>(p + 12));
  } else {
#pragma unroll
    for (int i = 0; i < 16; ++i) a[i] = bf_zero();
  }
  return a;
}
__device__ __forceinline__ v16bf load_a_frag_hard(const float* p, bool valid) {
  v16bf a;
  if (valid) {
    const float4 f0 = *reinterpret_cast<const float4*>(p);
    const float4 f1 = *reinterpret_cast<const float4*>(p + 4);
    const float4 f2 = *reinterpret_cast<const float4*>(p + 8);
    const float4 f3 = *reinterpret_cast<const float4*>(p + 12);
    a[0]=bf_hard(f0.x);  a[1]=bf_hard(f0.y);  a[2]=bf_hard(f0.z);  a[3]=bf_hard(f0.w);
    a[4]=bf_hard(f1.x);  a[5]=bf_hard(f1.y);  a[6]=bf_hard(f1.z);  a[7]=bf_hard(f1.w);
    a[8]=bf_hard(f2.x);  a[9]=bf_hard(f2.y);  a[10]=bf_hard(f2.z); a[11]=bf_hard(f2.w);
    a[12]=bf_hard(f3.x); a[13]=bf_hard(f3.y); a[14]=bf_hard(f3.z); a[15]=bf_hard(f3.w);
  } else {
#pragma unroll
    for (int i = 0; i < 16; ++i) a[i] = bf_zero();
  }
  return a;
}
// B-fragment (32x16 bf16): lane half h holds K = h*16 + e (16 consecutive).
// p points at (k0 + half*16) within a K-contiguous row.
__device__ __forceinline__ v16bf load_b_frag(const float* p, bool valid) {
  v16bf b;
  if (valid) {
    cvt4(b, 0,  *reinterpret_cast<const float4*>(p));
    cvt4(b, 4,  *reinterpret_cast<const float4*>(p + 4));
    cvt4(b, 8,  *reinterpret_cast<const float4*>(p + 8));
    cvt4(b, 12, *reinterpret_cast<const float4*>(p + 12));
  } else {
#pragma unroll
    for (int i = 0; i < 16; ++i) b[i] = bf_zero();
  }
  return b;
}
#define WMMA_BF16(a, b, c) \
  __builtin_amdgcn_wmma_f32_16x16x32_bf16(false, (a), false, (b), (short)0, (c), false, false)

// ---------------------------------------------------------------------------
// Kernel: zero fp32 buffer
// ---------------------------------------------------------------------------
__global__ void kuh_memset(float* p, int n) {
  int i = blockIdx.x * 256 + threadIdx.x;
  if (i < n) p[i] = 0.f;
}

// ---------------------------------------------------------------------------
// Kernel 1: x_feat[b,n,c] = sum_hw hard(mask[b,n,hw]) * x[b,c,hw]   (WMMA)
// One wave owns a 16(n) x 32(c) output block: A (hard) converted once,
// 2 WMMAs per 32-hw step.  grid = (BB*NT*CT2, KSPLIT), block = 32.
// ---------------------------------------------------------------------------
__global__ void kuh_pool_wmma(const float* __restrict__ x,
                              const float* __restrict__ mask,
                              float* __restrict__ xfeat) {
  int wid = blockIdx.x;
  int ct2 = wid % CT2; wid /= CT2;
  int nt  = wid % NT;  int b = wid / NT;
  const int lane = threadIdx.x, half = lane >> 4, lrow = lane & 15;

  const int n  = nt * 16 + lrow;
  const bool vn = n < NN;
  const float* mrow  = mask + (size_t)(b * NN + (vn ? n : 0)) * HWSZ;
  const float* xrow0 = x + (size_t)(b * CC + ct2 * 32 + lrow) * HWSZ;
  const float* xrow1 = xrow0 + (size_t)16 * HWSZ;

  v8f acc0 = {}, acc1 = {};
  const int hw_beg = blockIdx.y * KCH, hw_end = hw_beg + KCH;
  for (int hw0 = hw_beg; hw0 < hw_end; hw0 += 32) {
    v16bf a  = load_a_frag_hard(mrow + hw0 + half * 16, vn);
    v16bf b0 = load_b_frag(xrow0 + hw0 + half * 16, true);
    v16bf b1 = load_b_frag(xrow1 + hw0 + half * 16, true);
    acc0 = WMMA_BF16(a, b0, acc0);
    acc1 = WMMA_BF16(a, b1, acc1);
  }
#pragma unroll
  for (int v = 0; v < 8; ++v) {
    int row = nt * 16 + v + half * 8;
    if (row < NN) {
      float* dst = &xfeat[(size_t)(b * NN + row) * CC + ct2 * 32 + lrow];
      atomicAdd(dst,      acc0[v]);
      atomicAdd(dst + 16, acc1[v]);
    }
  }
}

// ---------------------------------------------------------------------------
// Kernel: generic Y[M,O] = X[M,K] @ W[O,K]^T (+bias) (+ReLU)   (WMMA)
// One wave owns a 16(m) x 32(o) block (A converted once, 2 WMMAs/step).
// grid = (ceil(M/16), ceil(O/32)), block = 32. K % 32 == 0.
// ---------------------------------------------------------------------------
__global__ void kuh_linear_wmma(const float* __restrict__ X,
                                const float* __restrict__ W,
                                const float* __restrict__ bias,
                                float* __restrict__ Y,
                                int M, int K, int O, int act) {
  const int mt = blockIdx.x, ot2 = blockIdx.y;
  const int lane = threadIdx.x, half = lane >> 4, lrow = lane & 15;

  const int m  = mt * 16 + lrow;       const bool vm  = m < M;
  const int o0 = ot2 * 32 + lrow;      const bool vo0 = o0 < O;
  const int o1 = o0 + 16;              const bool vo1 = o1 < O;
  const float* xrow  = X + (size_t)(vm  ? m  : 0) * K;
  const float* wrow0 = W + (size_t)(vo0 ? o0 : 0) * K;
  const float* wrow1 = W + (size_t)(vo1 ? o1 : 0) * K;

  v8f acc0 = {}, acc1 = {};
  for (int k0 = 0; k0 < K; k0 += 32) {
    v16bf a  = load_a_frag(xrow  + k0 + half * 16, vm);
    v16bf b0 = load_b_frag(wrow0 + k0 + half * 16, vo0);
    v16bf b1 = load_b_frag(wrow1 + k0 + half * 16, vo1);
    acc0 = WMMA_BF16(a, b0, acc0);
    acc1 = WMMA_BF16(a, b1, acc1);
  }
#pragma unroll
  for (int v = 0; v < 8; ++v) {
    int row = mt * 16 + v + half * 8;
    if (row < M) {
#pragma unroll
      for (int u = 0; u < 2; ++u) {
        int col = ot2 * 32 + u * 16 + lrow;
        if (col < O) {
          float r = (u ? acc1[v] : acc0[v]) + (bias ? bias[col] : 0.f);
          if (act == 1) r = fmaxf(r, 0.f);
          Y[(size_t)row * O + col] = r;
        }
      }
    }
  }
}

// ---------------------------------------------------------------------------
// Kernel: new_mask[b,n,hw] = sum_c kern[b,n,c] * x[b,c,hw]   (WMMA)
// One wave owns 16(n) x 32(hw); A (kernels) converted once per K-chunk.
// grid = (HWT2, BB*NT), block = 32
// ---------------------------------------------------------------------------
__global__ void kuh_maskpred_wmma(const float* __restrict__ kern,
                                  const float* __restrict__ x,
                                  float* __restrict__ out) {
  const int hwt2 = blockIdx.x;
  int wid = blockIdx.y;
  const int nt = wid % NT, b = wid / NT;
  const int lane = threadIdx.x, half = lane >> 4, lrow = lane & 15;

  const int n  = nt * 16 + lrow;
  const bool vn = n < NN;
  const float* krow = kern + (size_t)(b * NN + (vn ? n : 0)) * CC;
  const int hw0 = hwt2 * 32 + lrow;         // first hw column for this lane
  const float* xb0 = x + (size_t)b * CC * HWSZ + hw0;
  const float* xb1 = xb0 + 16;

  v8f acc0 = {}, acc1 = {};
  for (int c0 = 0; c0 < CC; c0 += 32) {
    v16bf a = load_a_frag(krow + c0 + half * 16, vn);
    const size_t coff = (size_t)(c0 + half * 16) * HWSZ;
    float t0[16], t1[16];
#pragma unroll
    for (int e = 0; e < 16; ++e) {
      t0[e] = xb0[coff + (size_t)e * HWSZ];
      t1[e] = xb1[coff + (size_t)e * HWSZ];
    }
    v16bf b0, b1;
#pragma unroll
    for (int e = 0; e < 16; e += 2) {
      cvt2(b0, e, t0[e], t0[e + 1]);
      cvt2(b1, e, t1[e], t1[e + 1]);
    }
    acc0 = WMMA_BF16(a, b0, acc0);
    acc1 = WMMA_BF16(a, b1, acc1);
  }
#pragma unroll
  for (int v = 0; v < 8; ++v) {
    int row = nt * 16 + v + half * 8;
    if (row < NN) {
      float* dst = &out[(size_t)(b * NN + row) * HWSZ + hw0];
      dst[0]  = acc0[v];
      dst[16] = acc1[v];
    }
  }
}

// ---------------------------------------------------------------------------
// Kernel: row LayerNorm over 256 cols, optional residual / sigmoid / relu.
// x read with (xstride, xoff) so halves of a [.,512] buffer can be used.
// grid = (#rows), block = 256.  mode: 0 none, 1 sigmoid, 2 relu.
// ---------------------------------------------------------------------------
__global__ void kuh_ln256(const float* __restrict__ x, int xstride, int xoff,
                          const float* __restrict__ res,
                          const float* __restrict__ w, const float* __restrict__ b,
                          float* __restrict__ y, int mode) {
  __shared__ float red[256];
  __shared__ float s_mean, s_rstd;
  const int row = blockIdx.x, tid = threadIdx.x;
  float v = x[(size_t)row * xstride + xoff + tid];
  if (res) v += res[(size_t)row * CC + tid];

  red[tid] = v; __syncthreads();
  for (int s = 128; s > 0; s >>= 1) { if (tid < s) red[tid] += red[tid + s]; __syncthreads(); }
  if (tid == 0) s_mean = red[0] * (1.f / 256.f);
  __syncthreads();
  float d = v - s_mean;
  red[tid] = d * d; __syncthreads();
  for (int s = 128; s > 0; s >>= 1) { if (tid < s) red[tid] += red[tid + s]; __syncthreads(); }
  if (tid == 0) s_rstd = rsqrtf(red[0] * (1.f / 256.f) + 1e-5f);
  __syncthreads();

  float r = d * s_rstd * w[tid] + b[tid];
  if (mode == 1)      r = 1.f / (1.f + __expf(-r));
  else if (mode == 2) r = fmaxf(r, 0.f);
  y[(size_t)row * CC + tid] = r;
}

// gate = i_in * p_in  (first 256 cols of the two [.,512] buffers)
__global__ void kuh_gate(const float* __restrict__ feats,
                         const float* __restrict__ params,
                         float* __restrict__ gate) {
  const int row = blockIdx.x, tid = threadIdx.x;
  gate[(size_t)row * CC + tid] =
      feats[(size_t)row * 2 * CC + tid] * params[(size_t)row * 2 * CC + tid];
}

// f = ug * p_out + ig * i_out
__global__ void kuh_combine(const float* __restrict__ ug, const float* __restrict__ po,
                            const float* __restrict__ ig, const float* __restrict__ io,
                            float* __restrict__ f) {
  const size_t i = (size_t)blockIdx.x * 256 + threadIdx.x;
  f[i] = ug[i] * po[i] + ig[i] * io[i];
}

__global__ void kuh_copy(const float* __restrict__ src, float* __restrict__ dst, int n) {
  int i = blockIdx.x * 256 + threadIdx.x;
  if (i < n) dst[i] = src[i];
}

// ---------------------------------------------------------------------------
// Kernel: MHA for one (b, h, s) per block.  d=32, softmax over 100 keys.
// qkv rows: [b*N+n, 768] = [q | k | v].  o rows: [b*N+s, 256].
// grid = BB*NHH*NN, block = 128
// ---------------------------------------------------------------------------
__global__ void kuh_attn(const float* __restrict__ qkv, float* __restrict__ o) {
  __shared__ float qs[DH], lg[NN], red[128];
  __shared__ float s_max, s_sum;
  int idx = blockIdx.x;
  const int sq = idx % NN; idx /= NN;
  const int h  = idx % NHH; const int b = idx / NHH;
  const int tid = threadIdx.x;
  const float scale = 0.17677669529663687f;   // 1/sqrt(32)

  if (tid < DH)
    qs[tid] = qkv[(size_t)(b * NN + sq) * (3 * CC) + h * DH + tid] * scale;
  __syncthreads();

  if (tid < NN) {
    const float* krow = qkv + (size_t)(b * NN + tid) * (3 * CC) + CC + h * DH;
    float acc = 0.f;
#pragma unroll
    for (int d = 0; d < DH; ++d) acc += qs[d] * krow[d];
    lg[tid] = acc;
  }
  __syncthreads();

  red[tid] = (tid < NN) ? lg[tid] : -3.4e38f;
  __syncthreads();
  for (int s = 64; s > 0; s >>= 1) { if (tid < s) red[tid] = fmaxf(red[tid], red[tid + s]); __syncthreads(); }
  if (tid == 0) s_max = red[0];
  __syncthreads();

  float e = (tid < NN) ? __expf(lg[tid] - s_max) : 0.f;
  if (tid < NN) lg[tid] = e;
  red[tid] = e; __syncthreads();
  for (int s = 64; s > 0; s >>= 1) { if (tid < s) red[tid] += red[tid + s]; __syncthreads(); }
  if (tid == 0) s_sum = red[0];
  __syncthreads();

  if (tid < DH) {
    const float inv = 1.f / s_sum;
    float acc = 0.f;
    for (int t = 0; t < NN; ++t)
      acc += lg[t] * qkv[(size_t)(b * NN + t) * (3 * CC) + 2 * CC + h * DH + tid];
    o[(size_t)(b * NN + sq) * CC + h * DH + tid] = acc * inv;
  }
}

// ---------------------------------------------------------------------------
// Host launcher
// ---------------------------------------------------------------------------
extern "C" void kernel_launch(void* const* d_in, const int* in_sizes, int n_in,
                              void* d_out, int out_size, void* d_ws, size_t ws_size,
                              hipStream_t stream) {
  const float* x         = (const float*)d_in[0];
  const float* proposal  = (const float*)d_in[1];
  const float* mask      = (const float*)d_in[2];
  const float* ku_dyn_w  = (const float*)d_in[3];
  const float* ku_dyn_b  = (const float*)d_in[4];
  const float* ku_in_w   = (const float*)d_in[5];
  const float* ku_in_b   = (const float*)d_in[6];
  const float* ku_ig_w   = (const float*)d_in[7];
  const float* ku_ig_b   = (const float*)d_in[8];
  const float* ku_ug_w   = (const float*)d_in[9];
  const float* ku_ug_b   = (const float*)d_in[10];
  const float* ku_nin_w  = (const float*)d_in[11];
  const float* ku_nin_b  = (const float*)d_in[12];
  const float* ku_nout_w = (const float*)d_in[13];
  const float* ku_nout_b = (const float*)d_in[14];
  const float* ku_inin_w = (const float*)d_in[15];
  const float* ku_inin_b = (const float*)d_in[16];
  const float* ku_inout_w= (const float*)d_in[17];
  const float* ku_inout_b= (const float*)d_in[18];
  const float* ku_fc_w   = (const float*)d_in[19];
  const float* ku_fc_b   = (const float*)d_in[20];
  const float* ku_fcn_w  = (const float*)d_in[21];
  const float* ku_fcn_b  = (const float*)d_in[22];
  const float* attn_in_w = (const float*)d_in[23];
  const float* attn_in_b = (const float*)d_in[24];
  const float* attn_out_w= (const float*)d_in[25];
  const float* attn_out_b= (const float*)d_in[26];
  const float* attn_ln_w = (const float*)d_in[27];
  const float* attn_ln_b = (const float*)d_in[28];
  const float* ffn_w1    = (const float*)d_in[29];
  const float* ffn_b1    = (const float*)d_in[30];
  const float* ffn_w2    = (const float*)d_in[31];
  const float* ffn_b2    = (const float*)d_in[32];
  const float* ffn_ln_w  = (const float*)d_in[33];
  const float* ffn_ln_b  = (const float*)d_in[34];
  const float* cls_fc_w  = (const float*)d_in[35];
  const float* cls_ln_w  = (const float*)d_in[36];
  const float* cls_ln_b  = (const float*)d_in[37];
  const float* fc_cls_w  = (const float*)d_in[38];
  const float* fc_cls_b  = (const float*)d_in[39];
  const float* mask_fc_w = (const float*)d_in[40];
  const float* mask_ln_w = (const float*)d_in[41];
  const float* mask_ln_b = (const float*)d_in[42];
  const float* fc_mask_w = (const float*)d_in[43];
  const float* fc_mask_b = (const float*)d_in[44];

  // workspace layout (floats); ~6.35 MB total
  float* ws = (float*)d_ws;
  float* x_feat = ws;              // 51200
  float* params = ws + 51200;      // 102400  [p_in | p_out]
  float* feats  = ws + 153600;     // 102400  [i_in | i_out]
  float* gate   = ws + 256000;     // 51200
  float* tmpA   = ws + 307200;     // 51200
  float* tmpB   = ws + 358400;     // 51200
  float* ig     = ws + 409600;     // 51200
  float* ug     = ws + 460800;     // 51200
  float* pout   = ws + 512000;     // 51200
  float* iout   = ws + 563200;     // 51200
  float* fbuf   = ws + 614400;     // 51200
  float* f2     = ws + 665600;     // 51200
  float* qkv    = ws + 716800;     // 153600
  float* obuf   = ws + 870400;     // 51200
  float* obj    = ws + 921600;     // 51200
  float* ffn_h  = ws + 972800;     // 409600
  float* obj2   = ws + 1382400;    // 51200
  float* clsf   = ws + 1433600;    // 51200
  float* kern   = ws + 1484800;    // 51200
  float* mfA    = ws + 1536000;    // 51200

  float* out_cls  = (float*)d_out;                       // B*N*NC   = 26600
  float* out_mask = out_cls + (size_t)BB * NN * NCL;     // B*N*H*W  = 8000000
  float* out_obj  = out_mask + (size_t)BB * NN * HWSZ;   // B*N*C    = 51200

  // ---- Stage A: hard-mask pooling (WMMA + fp32 atomics) -------------------
  kuh_memset<<<dim3((51200 + 255) / 256), dim3(256), 0, stream>>>(x_feat, 51200);
  kuh_pool_wmma<<<dim3(BB * NT * CT2, KSPLIT), dim3(32), 0, stream>>>(x, mask, x_feat);

  // ---- Kernel-update head (all GEMMs via WMMA) ----------------------------
  kuh_linear_wmma<<<dim3(13, 16), dim3(32), 0, stream>>>(x_feat,   ku_dyn_w, ku_dyn_b, params, BN, CC, 2 * CC, 0);
  kuh_linear_wmma<<<dim3(13, 16), dim3(32), 0, stream>>>(proposal, ku_in_w,  ku_in_b,  feats,  BN, CC, 2 * CC, 0);
  kuh_gate<<<dim3(BN), dim3(256), 0, stream>>>(feats, params, gate);

  kuh_linear_wmma<<<dim3(13, 8), dim3(32), 0, stream>>>(gate, ku_ig_w, ku_ig_b, tmpA, BN, CC, CC, 0);
  kuh_ln256<<<dim3(BN), dim3(256), 0, stream>>>(tmpA, CC, 0, nullptr, ku_inin_w, ku_inin_b, ig, 1);
  kuh_linear_wmma<<<dim3(13, 8), dim3(32), 0, stream>>>(gate, ku_ug_w, ku_ug_b, tmpB, BN, CC, CC, 0);
  kuh_ln256<<<dim3(BN), dim3(256), 0, stream>>>(tmpB, CC, 0, nullptr, ku_nin_w, ku_nin_b, ug, 1);

  kuh_ln256<<<dim3(BN), dim3(256), 0, stream>>>(params, 2 * CC, CC, nullptr, ku_nout_w,  ku_nout_b,  pout, 0);
  kuh_ln256<<<dim3(BN), dim3(256), 0, stream>>>(feats,  2 * CC, CC, nullptr, ku_inout_w, ku_inout_b, iout, 0);
  kuh_combine<<<dim3(BN), dim3(256), 0, stream>>>(ug, pout, ig, iout, fbuf);

  kuh_linear_wmma<<<dim3(13, 8), dim3(32), 0, stream>>>(fbuf, ku_fc_w, ku_fc_b, tmpA, BN, CC, CC, 0);
  kuh_ln256<<<dim3(BN), dim3(256), 0, stream>>>(tmpA, CC, 0, nullptr, ku_fcn_w, ku_fcn_b, f2, 2);

  // ---- Multi-head attention ----------------------------------------------
  kuh_linear_wmma<<<dim3(13, 24), dim3(32), 0, stream>>>(f2, attn_in_w, attn_in_b, qkv, BN, CC, 3 * CC, 0);
  kuh_attn<<<dim3(BB * NHH * NN), dim3(128), 0, stream>>>(qkv, obuf);
  kuh_linear_wmma<<<dim3(13, 8), dim3(32), 0, stream>>>(obuf, attn_out_w, attn_out_b, tmpA, BN, CC, CC, 0);
  kuh_ln256<<<dim3(BN), dim3(256), 0, stream>>>(tmpA, CC, 0, f2, attn_ln_w, attn_ln_b, obj, 0);

  // ---- FFN ----------------------------------------------------------------
  kuh_linear_wmma<<<dim3(13, 64), dim3(32), 0, stream>>>(obj, ffn_w1, ffn_b1, ffn_h, BN, CC, FFD, 1);
  kuh_linear_wmma<<<dim3(13, 8), dim3(32), 0, stream>>>(ffn_h, ffn_w2, ffn_b2, tmpA, BN, FFD, CC, 0);
  kuh_ln256<<<dim3(BN), dim3(256), 0, stream>>>(tmpA, CC, 0, obj, ffn_ln_w, ffn_ln_b, obj2, 0);

  // ---- Class head ---------------------------------------------------------
  kuh_linear_wmma<<<dim3(13, 8), dim3(32), 0, stream>>>(obj2, cls_fc_w, nullptr, tmpA, BN, CC, CC, 0);
  kuh_ln256<<<dim3(BN), dim3(256), 0, stream>>>(tmpA, CC, 0, nullptr, cls_ln_w, cls_ln_b, clsf, 2);
  kuh_linear_wmma<<<dim3(13, 5), dim3(32), 0, stream>>>(clsf, fc_cls_w, fc_cls_b, out_cls, BN, CC, NCL, 0);

  // ---- Mask head ----------------------------------------------------------
  const float* mf_in = obj2;
  float* mf_out = mfA;
  for (int i = 0; i < 3; ++i) {
    kuh_linear_wmma<<<dim3(13, 8), dim3(32), 0, stream>>>(mf_in, mask_fc_w + (size_t)i * CC * CC,
                                                          nullptr, tmpA, BN, CC, CC, 0);
    kuh_ln256<<<dim3(BN), dim3(256), 0, stream>>>(tmpA, CC, 0, nullptr,
                                                  mask_ln_w + (size_t)i * CC,
                                                  mask_ln_b + (size_t)i * CC, mf_out, 2);
    mf_in = mf_out;
    mf_out = (i == 0) ? tmpB : mfA;   // ping-pong
  }
  kuh_linear_wmma<<<dim3(13, 8), dim3(32), 0, stream>>>(mf_in, fc_mask_w, fc_mask_b, kern, BN, CC, CC, 0);

  // ---- Outputs: new mask predictions (WMMA) + obj passthrough -------------
  kuh_maskpred_wmma<<<dim3(HWT2, BB * NT), dim3(32), 0, stream>>>(kern, x, out_mask);
  kuh_copy<<<dim3((51200 + 255) / 256), dim3(256), 0, stream>>>(obj2, out_obj, 51200);
}
